// MEFL_43791486550521
// MI455X (gfx1250) — matrix operand order repaired and checked by reading, not compiled, not run on hardware
//
// MI455X (gfx1250) implementation: bf16 WMMA (v_wmma_f32_16x16x32_bf16) for all
// GEMM-like stages; fully fused ARM-attention + edge projection so the 292MB
// `fe` tensor never touches HBM. wave32, 256-thread (8-wave) blocks.
#include <hip/hip_runtime.h>
#include <math.h>

typedef __bf16 bf16_t;
typedef __attribute__((ext_vector_type(16))) __bf16 v16bf;
typedef __attribute__((ext_vector_type(8)))  float  v8f;

#define WMMA_BF16(a, b, c) \
  __builtin_amdgcn_wmma_f32_16x16x32_bf16(false, (a), false, (b), (short)0, (c), false, false)

__device__ __forceinline__ v8f v8f_zero() {
  v8f z;
#pragma unroll
  for (int i = 0; i < 8; ++i) z[i] = 0.f;
  return z;
}

union FragU { v16bf v; uint4 q[2]; };

// A fragment (16x32 bf16, M x K), row-major source with leading dim ld.
// lanes 0-15: K 0..7 & 16..23 ; lanes 16-31: K 8..15 & 24..31 (ISA 7.12.2)
__device__ __forceinline__ v16bf load_a(const bf16_t* base, int ld) {
  int lane = threadIdx.x & 31;
  int hf = lane >> 4, m = lane & 15;
  const bf16_t* p = base + (size_t)m * ld + 8 * hf;
  FragU f;
  f.q[0] = *(const uint4*)(p);
  f.q[1] = *(const uint4*)(p + 16);
  return f.v;
}

// B fragment (32x16 bf16, K x N) sourced from a row-major [N,K] matrix
// (torch Linear weight layout): lane n = lane%16, K = 8*half + 0..7 & 16 + 8*half + 0..7.
__device__ __forceinline__ v16bf load_b(const bf16_t* base, int ld) {
  int lane = threadIdx.x & 31;
  int hf = lane >> 4, n = lane & 15;
  const bf16_t* p = base + (size_t)n * ld + 8 * hf;
  FragU f;
  f.q[0] = *(const uint4*)(p);
  f.q[1] = *(const uint4*)(p + 16);
  return f.v;
}

// D layout: element r -> row m = r + 8*half, col n = lane%16
__device__ __forceinline__ void wstore_f32(float* base, int ld, v8f acc, float scale, float bias) {
  int lane = threadIdx.x & 31, n = lane & 15, hf = lane >> 4;
#pragma unroll
  for (int r = 0; r < 8; ++r) base[(size_t)(r + 8 * hf) * ld + n] = acc[r] * scale + bias;
}
__device__ __forceinline__ void wstore_bf16(bf16_t* base, int ld, v8f acc, float bias) {
  int lane = threadIdx.x & 31, n = lane & 15, hf = lane >> 4;
#pragma unroll
  for (int r = 0; r < 8; ++r) base[(size_t)(r + 8 * hf) * ld + n] = (bf16_t)(acc[r] + bias);
}

__constant__ int c_subidx[14] = {0, 0, 1, 1, 2, 2, 4, 4, 7, 7, 8, 8, 11, 11};

// ---------------- small utility kernels ----------------
__global__ __launch_bounds__(256) void k_f2h(const float* __restrict__ s, bf16_t* __restrict__ d, int n) {
  int i = blockIdx.x * 256 + threadIdx.x;
  if (i < n) d[i] = (bf16_t)s[i];
}

// x [4,49,512] f32 -> xh [4,64,512] bf16 with zero pad rows (d>=49)
__global__ __launch_bounds__(256) void k_xpad(const float* __restrict__ x, bf16_t* __restrict__ xh) {
  int idx = blockIdx.x * 256 + threadIdx.x;   // < 4*64*512
  int c = idx & 511, d = (idx >> 9) & 63, b = idx >> 15;
  xh[idx] = (d < 49) ? (bf16_t)x[((b * 49 + d) << 9) + c] : (bf16_t)0.f;
}

// f_v [108,512] f32 -> fvh [112,512] bf16, zero pad rows
__global__ __launch_bounds__(256) void k_fvh(const float* __restrict__ fv, bf16_t* __restrict__ out) {
  int idx = blockIdx.x * 256 + threadIdx.x;   // < 112*512
  int row = idx >> 9;
  out[idx] = (row < 108) ? (bf16_t)fv[idx] : (bf16_t)0.f;
}

// ---------------- generic batched bf16 WMMA GEMM ----------------
// out[m,n] = sum_k A[m,k] * W[n,k] + bias[n].  grid=(M/16, N/128, Z); 8 waves.
// mode 0: f32 row-major; 1: bf16 row-major; 2: bf16 block-transposed
//         (out[(m>>6)*ldo + n)*64 + (m&63)], 64-row token blocks).
__global__ __launch_bounds__(256) void k_gemm(
    const bf16_t* __restrict__ A, int lda, long sA,
    const bf16_t* __restrict__ W, int ldw, long sW,
    const float* __restrict__ bias,
    float* __restrict__ outF, bf16_t* __restrict__ outH,
    int ldo, long sO, int M, int K, int mode) {
  int z = blockIdx.z;
  A += (size_t)z * sA;
  W += (size_t)z * sW;
  int tid = threadIdx.x, w = tid >> 5, lane = tid & 31;
  int m0 = blockIdx.x * 16;
  int n0 = (blockIdx.y * 8 + w) * 16;
  const bf16_t* Ap = A + (size_t)m0 * lda;
  const bf16_t* Wp = W + (size_t)n0 * ldw;
  v8f acc = v8f_zero();
  for (int k0 = 0; k0 < K; k0 += 32) {
    v16bf af = load_a(Ap + k0, lda);
    v16bf bf = load_b(Wp + k0, ldw);
    if (k0 + 128 < K) __builtin_prefetch(Wp + (size_t)(lane & 15) * ldw + k0 + 128, 0, 1);
    acc = WMMA_BF16(af, bf, acc);
  }
  int n = n0 + (lane & 15), hf = lane >> 4;
  float bv = bias ? bias[n] : 0.f;
#pragma unroll
  for (int r = 0; r < 8; ++r) {
    int m = m0 + r + 8 * hf;
    if (m < M) {
      float v = acc[r] + bv;
      if (mode == 0)      outF[(size_t)z * sO + (size_t)m * ldo + n] = v;
      else if (mode == 1) outH[(size_t)z * sO + (size_t)m * ldo + n] = (bf16_t)v;
      else                outH[(size_t)z * sO + ((size_t)(m >> 6) * ldo + n) * 64 + (m & 63)] = (bf16_t)v;
    }
  }
}

// ---------------- node-block BN (stats over b,d per (n,c); fc bias cancels) --
__global__ __launch_bounds__(256) void k_bn_stats(const bf16_t* __restrict__ fu,
                                                  float* __restrict__ bnm, float* __restrict__ bnr) {
  int n = blockIdx.x;
  for (int c = threadIdx.x; c < 512; c += 256) {
    float s = 0.f, q = 0.f;
    const bf16_t* base = fu + (size_t)n * 256 * 512 + c;
    for (int b = 0; b < 4; ++b)
      for (int d = 0; d < 49; ++d) {
        float v = (float)base[(size_t)(b * 64 + d) * 512];
        s += v; q += v * v;
      }
    float m = s * (1.f / 196.f);
    float var = q * (1.f / 196.f) - m * m;
    bnm[n * 512 + c] = m;
    bnr[n * 512 + c] = rsqrtf(var + 1e-5f);
  }
}

__global__ __launch_bounds__(256) void k_bn_apply(bf16_t* __restrict__ fu,
                                                  const float* __restrict__ bnm, const float* __restrict__ bnr,
                                                  const float* __restrict__ gw, const float* __restrict__ gb) {
  long idx = (long)blockIdx.x * 256 + threadIdx.x;   // < 27*256*512
  int c = (int)(idx & 511);
  int row = (int)((idx >> 9) & 255);
  int n = (int)(idx >> 17);
  int d = row & 63;
  float o = 0.f;
  if (d < 49) {
    float h = (float)fu[idx];
    int nc = n * 512 + c;
    o = fmaxf((h - bnm[nc]) * bnr[nc] * gw[nc] + gb[nc], 0.f);
  }
  fu[idx] = (bf16_t)o;
}

__global__ __launch_bounds__(256) void k_fv_gap(const bf16_t* __restrict__ fu, float* __restrict__ fv) {
  int idx = blockIdx.x * 256 + threadIdx.x;   // < 108*512
  int c = idx & 511, bn = idx >> 9;
  int b = bn / 27, n = bn % 27;
  const bf16_t* base = fu + ((size_t)n * 256 + b * 64) * 512 + c;
  float s = 0.f;
  for (int d = 0; d < 49; ++d) s += (float)base[(size_t)d * 512];
  fv[idx] = s * (1.f / 49.f);
}

// ---------------- FAM: fused q-GEMM + cross-attention, one block per (b,n) --
__global__ __launch_bounds__(256) void k_fam(const bf16_t* __restrict__ fu,
                                             const bf16_t* __restrict__ qW, const float* __restrict__ qb,
                                             const bf16_t* __restrict__ kx, const bf16_t* __restrict__ vxT,
                                             bf16_t* __restrict__ feat) {
  __shared__ __align__(16) bf16_t qs[64 * 256];
  __shared__ __align__(16) float  S[64 * 64];
  __shared__ __align__(16) bf16_t att[64 * 64];
  int bn = blockIdx.x, b = bn / 27, n = bn % 27;
  int tid = threadIdx.x, w = tid >> 5, lane = tid & 31;
  const bf16_t* A = fu + ((size_t)n * 256 + b * 64) * 512;
  // q = f_u @ fam_qW.T + qb  (64x256)
  for (int idx = w; idx < 64; idx += 8) {
    int mt = idx & 3, nt = idx >> 2;
    v8f acc = v8f_zero();
    for (int k0 = 0; k0 < 512; k0 += 32) {
      v16bf af = load_a(A + (size_t)mt * 16 * 512 + k0, 512);
      v16bf bf = load_b(qW + (size_t)nt * 16 * 512 + k0, 512);
      acc = WMMA_BF16(af, bf, acc);
    }
    wstore_bf16(qs + (size_t)(mt * 16) * 256 + nt * 16, 256, acc, qb[nt * 16 + (lane & 15)]);
  }
  __syncthreads();
  // S = q @ k[b]^T * scale (64x64)
  const bf16_t* kxb = kx + (size_t)b * 64 * 256;
  for (int idx = w; idx < 16; idx += 8) {
    int mt = idx & 3, nt = idx >> 2;
    v8f acc = v8f_zero();
    for (int k0 = 0; k0 < 256; k0 += 32) {
      v16bf af = load_a(qs + (size_t)mt * 16 * 256 + k0, 256);
      v16bf bf = load_b(kxb + (size_t)nt * 16 * 256 + k0, 256);
      acc = WMMA_BF16(af, bf, acc);
    }
    wstore_f32(S + (size_t)(mt * 16) * 64 + nt * 16, 64, acc, 0.0625f, 0.f);
  }
  __syncthreads();
  if (tid < 64) {
    bf16_t* arow = att + tid * 64;
    if (tid < 49) {
      const float* row = S + tid * 64;
      float mx = -1e30f;
      for (int e2 = 0; e2 < 49; ++e2) mx = fmaxf(mx, row[e2]);
      float se = 0.f;
      for (int e2 = 0; e2 < 49; ++e2) se += __expf(row[e2] - mx);
      float inv = 1.f / se;
      for (int e2 = 0; e2 < 49; ++e2) arow[e2] = (bf16_t)(__expf(row[e2] - mx) * inv);
      for (int e2 = 49; e2 < 64; ++e2) arow[e2] = (bf16_t)0.f;
    } else {
      for (int e2 = 0; e2 < 64; ++e2) arow[e2] = (bf16_t)0.f;
    }
  }
  __syncthreads();
  // feat = att @ v[b]  (64x512); pad rows naturally zero
  const bf16_t* vt = vxT + (size_t)b * 512 * 64;
  bf16_t* fb = feat + (size_t)bn * 64 * 512;
  for (int idx = w; idx < 128; idx += 8) {
    int mt = idx & 3, ct = idx >> 2;
    v8f acc = v8f_zero();
    for (int k0 = 0; k0 < 64; k0 += 32) {
      v16bf af = load_a(att + (size_t)mt * 16 * 64 + k0, 64);
      v16bf bf = load_b(vt + (size_t)(ct * 16) * 64 + k0, 64);
      acc = WMMA_BF16(af, bf, acc);
    }
    wstore_bf16(fb + (size_t)(mt * 16) * 512 + ct * 16, 512, acc, 0.f);
  }
}

// ---------------- fused ARM attention + edge projection + token-mean --------
// One block per (edge e=i*27+j, batch b). Never spills `fe` to HBM: emits only
// per-edge column means (tb) and scalar sum/sumsq partials (st) for the gem-BN.
__global__ __launch_bounds__(256) void k_arm_edge(const bf16_t* __restrict__ Qb, const bf16_t* __restrict__ Kb,
                                                  const bf16_t* __restrict__ VTb,
                                                  const bf16_t* __restrict__ epw, const float* __restrict__ epb,
                                                  float* __restrict__ tb, float* __restrict__ st) {
  __shared__ __align__(16) float  S[64 * 64];
  __shared__ __align__(16) bf16_t att[64 * 64];
  __shared__ __align__(16) bf16_t fes[64 * 256];
  __shared__ float red[256];
  int e = blockIdx.x, b = blockIdx.y;
  int i = e / 27, j = e % 27;
  int tid = threadIdx.x, w = tid >> 5, lane = tid & 31;
  const bf16_t* Qp = Qb + (size_t)(b * 27 + j) * 64 * 256;
  const bf16_t* Kp = Kb + (size_t)(b * 27 + i) * 64 * 256;
  const bf16_t* VT = VTb + (size_t)(b * 27 + i) * 512 * 64;
  // scores
  for (int idx = w; idx < 16; idx += 8) {
    int mt = idx & 3, nt = idx >> 2;
    v8f acc = v8f_zero();
    for (int k0 = 0; k0 < 256; k0 += 32) {
      v16bf af = load_a(Qp + (size_t)mt * 16 * 256 + k0, 256);
      v16bf bf = load_b(Kp + (size_t)nt * 16 * 256 + k0, 256);
      acc = WMMA_BF16(af, bf, acc);
    }
    wstore_f32(S + (size_t)(mt * 16) * 64 + nt * 16, 64, acc, 0.0625f, 0.f);
  }
  __syncthreads();
  if (tid < 64) {
    bf16_t* arow = att + tid * 64;
    if (tid < 49) {
      const float* row = S + tid * 64;
      float mx = -1e30f;
      for (int e2 = 0; e2 < 49; ++e2) mx = fmaxf(mx, row[e2]);
      float se = 0.f;
      for (int e2 = 0; e2 < 49; ++e2) se += __expf(row[e2] - mx);
      float inv = 1.f / se;
      for (int e2 = 0; e2 < 49; ++e2) arow[e2] = (bf16_t)(__expf(row[e2] - mx) * inv);
      for (int e2 = 49; e2 < 64; ++e2) arow[e2] = (bf16_t)0.f;
    } else {
      for (int e2 = 0; e2 < 64; ++e2) arow[e2] = (bf16_t)0.f;
    }
  }
  __syncthreads();
  // y = (att @ V) @ ep_W.T, K split into two 256-wide fe phases;
  // each wave holds 16 persistent f32 accumulators (its 4 mt x 4 nt tiles).
  v8f y[16];
#pragma unroll
  for (int t2 = 0; t2 < 16; ++t2) y[t2] = v8f_zero();
  for (int ph = 0; ph < 2; ++ph) {
    __syncthreads();
    for (int idx = w; idx < 64; idx += 8) {
      int mt = idx & 3, ct = idx >> 2;
      v8f acc = v8f_zero();
      for (int k0 = 0; k0 < 64; k0 += 32) {
        v16bf af = load_a(att + (size_t)mt * 16 * 64 + k0, 64);
        v16bf bf = load_b(VT + (size_t)(ph * 256 + ct * 16) * 64 + k0, 64);
        acc = WMMA_BF16(af, bf, acc);
      }
      wstore_bf16(fes + (size_t)(mt * 16) * 256 + ct * 16, 256, acc, 0.f);
    }
    __syncthreads();
    for (int tt = 0; tt < 16; ++tt) {
      int mt = tt & 3, ntl = tt >> 2;
      int n0 = (w * 4 + ntl) * 16;
      v8f acc = y[tt];
      for (int k0 = 0; k0 < 256; k0 += 32) {
        v16bf af = load_a(fes + (size_t)mt * 16 * 256 + k0, 256);
        v16bf bf = load_b(epw + (size_t)n0 * 512 + ph * 256 + k0, 512);
        acc = WMMA_BF16(af, bf, acc);
      }
      y[tt] = acc;
    }
  }
  // deterministic reductions: column sums over d<49 (wave-exclusive columns)
  float s1 = 0.f, s2 = 0.f;
  int nl = lane & 15, hf = lane >> 4;
  for (int ntl = 0; ntl < 4; ++ntl) {
    int n = (w * 4 + ntl) * 16 + nl;
    float bv = epb[n];
    float cs = 0.f;
    for (int mt = 0; mt < 4; ++mt) {
      v8f a = y[ntl * 4 + mt];
#pragma unroll
      for (int r = 0; r < 8; ++r) {
        int m = mt * 16 + r + 8 * hf;
        if (m < 49) { float v = a[r] + bv; cs += v; s1 += v; s2 += v * v; }
      }
    }
    cs += __shfl_xor(cs, 16);
    if (hf == 0) tb[((size_t)b * 729 + e) * 512 + n] = cs * (1.f / 49.f);
  }
  red[tid] = s1; __syncthreads();
  for (int s = 128; s > 0; s >>= 1) { if (tid < s) red[tid] += red[tid + s]; __syncthreads(); }
  float S1 = red[0]; __syncthreads();
  red[tid] = s2; __syncthreads();
  for (int s = 128; s > 0; s >>= 1) { if (tid < s) red[tid] += red[tid + s]; __syncthreads(); }
  if (tid == 0) {
    st[((size_t)b * 729 + e) * 2 + 0] = S1;
    st[((size_t)b * 729 + e) * 2 + 1] = red[0];
  }
}

// gem BN (per-edge over b,d,c; affine commutes with token-mean) -> f_e f32+bf16
__global__ __launch_bounds__(256) void k_gem_bn(const float* __restrict__ tb, const float* __restrict__ st,
                                                const float* __restrict__ gw, const float* __restrict__ gb,
                                                float* __restrict__ fe, bf16_t* __restrict__ fe16) {
  int e = blockIdx.x, tid = threadIdx.x;
  if (e >= 729) {   // zero the 12 pad rows of fe16 (rows 2916..2927)
    for (int idx = tid; idx < 12 * 512; idx += 256) fe16[(size_t)2916 * 512 + idx] = (bf16_t)0.f;
    return;
  }
  float s1 = 0.f, s2 = 0.f;
  for (int b = 0; b < 4; ++b) {
    s1 += st[((size_t)b * 729 + e) * 2 + 0];
    s2 += st[((size_t)b * 729 + e) * 2 + 1];
  }
  const float inv_cnt = 1.f / (4.f * 49.f * 512.f);
  float m = s1 * inv_cnt;
  float var = s2 * inv_cnt - m * m;
  float rs = rsqrtf(var + 1e-5f);
  float wv = gw[e], bv = gb[e];
  for (int idx = tid; idx < 4 * 512; idx += 256) {
    int b = idx >> 9, c = idx & 511;
    size_t o = ((size_t)b * 729 + e) * 512 + c;
    float v = (tb[o] - m) * rs * wv + bv;
    fe[o] = v;
    fe16[o] = (bf16_t)v;
  }
}

// ---------------- GatedGNN elementwise stages -------------------------------
__global__ __launch_bounds__(256) void k_edge_upd(const float* __restrict__ vi, const float* __restrict__ vj,
                                                  const float* __restrict__ el,
                                                  const float* __restrict__ bw, const float* __restrict__ bb,
                                                  float* __restrict__ fe, bf16_t* __restrict__ fe16) {
  int e = blockIdx.x, i = e / 27, j = e % 27, tid = threadIdx.x;
  __shared__ float mb[2048];
  __shared__ float red[256];
  float s1 = 0.f, s2 = 0.f;
  for (int idx = tid; idx < 2048; idx += 256) {
    int b = idx >> 9, c = idx & 511;
    float v = vi[(b * 27 + i) * 512 + c] + vj[(b * 27 + j) * 512 + c] +
              el[((size_t)b * 729 + e) * 512 + c];
    mb[idx] = v; s1 += v; s2 += v * v;
  }
  red[tid] = s1; __syncthreads();
  for (int s = 128; s > 0; s >>= 1) { if (tid < s) red[tid] += red[tid + s]; __syncthreads(); }
  float S1 = red[0]; __syncthreads();
  red[tid] = s2; __syncthreads();
  for (int s = 128; s > 0; s >>= 1) { if (tid < s) red[tid] += red[tid + s]; __syncthreads(); }
  float S2 = red[0]; __syncthreads();
  float m = S1 * (1.f / 2048.f);
  float var = S2 * (1.f / 2048.f) - m * m;
  float rs = rsqrtf(var + 1e-5f);
  float wv = bw[e], bv = bb[e];
  for (int idx = tid; idx < 2048; idx += 256) {
    int b = idx >> 9, c = idx & 511;
    float r_ = fmaxf((mb[idx] - m) * rs * wv + bv, 0.f);
    size_t o = ((size_t)b * 729 + e) * 512 + c;
    float nv = fe[o] + r_;
    fe[o] = nv;
    fe16[o] = (bf16_t)nv;
  }
}

__global__ __launch_bounds__(256) void k_agg(const float* __restrict__ fe, const float* __restrict__ ui,
                                             const float* __restrict__ uj, float* __restrict__ agg) {
  int bi = blockIdx.x, b = bi / 27, i = bi % 27, tid = threadIdx.x;
  for (int c = tid; c < 512; c += 256) {
    float mx = -1e30f;
    for (int jj = 0; jj < 27; ++jj) {
      float s = 1.f / (1.f + __expf(-fe[((size_t)b * 729 + i * 27 + jj) * 512 + c]));
      mx = fmaxf(mx, s);
    }
    float se = 0.f, dot = 0.f;
    for (int jj = 0; jj < 27; ++jj) {
      float s = 1.f / (1.f + __expf(-fe[((size_t)b * 729 + i * 27 + jj) * 512 + c]));
      float ex = __expf(s - mx);
      se += ex;
      dot += ex * uj[(b * 27 + jj) * 512 + c];
    }
    agg[(b * 27 + i) * 512 + c] = ui[(b * 27 + i) * 512 + c] + dot / (se * 27.f);
  }
}

__global__ __launch_bounds__(256) void k_node_upd(const float* __restrict__ agg,
                                                  const float* __restrict__ bw, const float* __restrict__ bb,
                                                  float* __restrict__ fv) {
  int i = blockIdx.x, tid = threadIdx.x;
  __shared__ float red[256];
  float s1 = 0.f, s2 = 0.f;
  for (int idx = tid; idx < 2048; idx += 256) {
    int b = idx >> 9, c = idx & 511;
    float v = agg[(b * 27 + i) * 512 + c];
    s1 += v; s2 += v * v;
  }
  red[tid] = s1; __syncthreads();
  for (int s = 128; s > 0; s >>= 1) { if (tid < s) red[tid] += red[tid + s]; __syncthreads(); }
  float S1 = red[0]; __syncthreads();
  red[tid] = s2; __syncthreads();
  for (int s = 128; s > 0; s >>= 1) { if (tid < s) red[tid] += red[tid + s]; __syncthreads(); }
  float S2 = red[0]; __syncthreads();
  float m = S1 * (1.f / 2048.f);
  float var = S2 * (1.f / 2048.f) - m * m;
  float rs = rsqrtf(var + 1e-5f);
  float wv = bw[i], bv = bb[i];
  for (int idx = tid; idx < 2048; idx += 256) {
    int b = idx >> 9, c = idx & 511;
    int o = (b * 27 + i) * 512 + c;
    fv[o] = fv[o] + fmaxf((agg[o] - m) * rs * wv + bv, 0.f);
  }
}

// ---------------- SC cosine-similarity head ---------------------------------
__global__ __launch_bounds__(64) void k_sc(const float* __restrict__ fv, const float* __restrict__ msc,
                                           const float* __restrict__ ssc, float* __restrict__ out) {
  int id = blockIdx.x, b = id / 41, o = id % 41, t = threadIdx.x;
  __shared__ float ra[64], rb[64], rc[64];
  int node = (o < 27) ? o : c_subidx[o - 27];
  const float* anc = (o < 27) ? (msc + (size_t)o * 512) : (ssc + (size_t)(o - 27) * 512);
  const float* f = fv + (size_t)(b * 27 + node) * 512;
  float a = 0.f, bq = 0.f, cq = 0.f;
  for (int c = t; c < 512; c += 64) {
    float x = f[c];
    float an = fmaxf(anc[c], 0.f);
    a += x * x; bq += an * an; cq += x * an;
  }
  ra[t] = a; rb[t] = bq; rc[t] = cq; __syncthreads();
  for (int s = 32; s > 0; s >>= 1) {
    if (t < s) { ra[t] += ra[t + s]; rb[t] += rb[t + s]; rc[t] += rc[t + s]; }
    __syncthreads();
  }
  if (t == 0) {
    float nf = fmaxf(sqrtf(ra[0]), 1e-12f);
    float na = fmaxf(sqrtf(rb[0]), 1e-12f);
    out[b * 41 + o] = rc[0] / (nf * na);
  }
}

// ---------------- launch ----------------------------------------------------
extern "C" void kernel_launch(void* const* d_in, const int* in_sizes, int n_in,
                              void* d_out, int out_size, void* d_ws, size_t ws_size,
                              hipStream_t stream) {
  const float* x     = (const float*)d_in[0];
  const float* nbW   = (const float*)d_in[1];
  const float* nbbnw = (const float*)d_in[3];
  const float* nbbnb = (const float*)d_in[4];
  const float* famqW = (const float*)d_in[5];  const float* famqb = (const float*)d_in[6];
  const float* famkW = (const float*)d_in[7];  const float* famkb = (const float*)d_in[8];
  const float* famvW = (const float*)d_in[9];  const float* famvb = (const float*)d_in[10];
  const float* armqW = (const float*)d_in[11]; const float* armqb = (const float*)d_in[12];
  const float* armkW = (const float*)d_in[13]; const float* armkb = (const float*)d_in[14];
  const float* armvW = (const float*)d_in[15]; const float* armvb = (const float*)d_in[16];
  const float* epW   = (const float*)d_in[17]; const float* epb   = (const float*)d_in[18];
  const float* gemw  = (const float*)d_in[19]; const float* gemb  = (const float*)d_in[20];
  const float* guW   = (const float*)d_in[21];
  const float* gvW   = (const float*)d_in[22];
  const float* gaW   = (const float*)d_in[23];
  const float* gbW   = (const float*)d_in[24];
  const float* geW   = (const float*)d_in[25];
  const float* gbnvw = (const float*)d_in[26]; const float* gbnvb = (const float*)d_in[27];
  const float* gbnew = (const float*)d_in[28]; const float* gbneb = (const float*)d_in[29];
  const float* mansc = (const float*)d_in[30]; const float* subsc = (const float*)d_in[31];
  (void)in_sizes; (void)n_in; (void)out_size;

  size_t off = 0;
  auto alloc = [&](size_t elems, size_t esz) -> void* {
    void* p = (char*)d_ws + off;
    off += (elems * esz + 255) & ~(size_t)255;
    return p;
  };
  bf16_t* xh    = (bf16_t*)alloc(4 * 64 * 512, 2);
  bf16_t* nbWh  = (bf16_t*)alloc(27 * 512 * 512, 2);
  bf16_t* famqh = (bf16_t*)alloc(256 * 512, 2);
  bf16_t* famkh = (bf16_t*)alloc(256 * 512, 2);
  bf16_t* famvh = (bf16_t*)alloc(512 * 512, 2);
  bf16_t* armqh = (bf16_t*)alloc(256 * 512, 2);
  bf16_t* armkh = (bf16_t*)alloc(256 * 512, 2);
  bf16_t* armvh = (bf16_t*)alloc(512 * 512, 2);
  bf16_t* epwh  = (bf16_t*)alloc(512 * 512, 2);
  bf16_t* gnnuh = (bf16_t*)alloc(2 * 512 * 512, 2);
  bf16_t* gnnvh = (bf16_t*)alloc(2 * 512 * 512, 2);
  bf16_t* gnnah = (bf16_t*)alloc(2 * 512 * 512, 2);
  bf16_t* gnnbh = (bf16_t*)alloc(2 * 512 * 512, 2);
  bf16_t* gnneh = (bf16_t*)alloc(2 * 512 * 512, 2);
  bf16_t* fu    = (bf16_t*)alloc(27 * 256 * 512, 2);
  float*  bnm   = (float*)alloc(27 * 512, 4);
  float*  bnr   = (float*)alloc(27 * 512, 4);
  float*  fv    = (float*)alloc(108 * 512, 4);
  bf16_t* fvh   = (bf16_t*)alloc(112 * 512, 2);
  bf16_t* kx    = (bf16_t*)alloc(4 * 64 * 256, 2);
  bf16_t* vxT   = (bf16_t*)alloc(4 * 512 * 64, 2);
  bf16_t* feat  = (bf16_t*)alloc(108 * 64 * 512, 2);
  bf16_t* Qb    = (bf16_t*)alloc(108 * 64 * 256, 2);
  bf16_t* Kb    = (bf16_t*)alloc(108 * 64 * 256, 2);
  bf16_t* VTb   = (bf16_t*)alloc(108 * 512 * 64, 2);
  float*  tbuf  = (float*)alloc(4 * 729 * 512, 4);
  float*  stat  = (float*)alloc(4 * 729 * 2, 4);
  float*  fe    = (float*)alloc(4 * 729 * 512, 4);
  bf16_t* fe16  = (bf16_t*)alloc(2928 * 512, 2);
  float*  vi    = (float*)alloc(112 * 512, 4);
  float*  vj    = (float*)alloc(112 * 512, 4);
  float*  ui    = (float*)alloc(112 * 512, 4);
  float*  uj    = (float*)alloc(112 * 512, 4);
  float*  el    = (float*)alloc(2928 * 512, 4);
  float*  agg   = (float*)alloc(108 * 512, 4);
  if (ws_size < off) return;

  auto f2h = [&](const float* s, bf16_t* d, int n) {
    k_f2h<<<(n + 255) / 256, 256, 0, stream>>>(s, d, n);
  };
  f2h(nbW, nbWh, 27 * 512 * 512);
  f2h(famqW, famqh, 256 * 512);
  f2h(famkW, famkh, 256 * 512);
  f2h(famvW, famvh, 512 * 512);
  f2h(armqW, armqh, 256 * 512);
  f2h(armkW, armkh, 256 * 512);
  f2h(armvW, armvh, 512 * 512);
  f2h(epW, epwh, 512 * 512);
  f2h(guW, gnnuh, 2 * 512 * 512);
  f2h(gvW, gnnvh, 2 * 512 * 512);
  f2h(gaW, gnnah, 2 * 512 * 512);
  f2h(gbW, gnnbh, 2 * 512 * 512);
  f2h(geW, gnneh, 2 * 512 * 512);
  k_xpad<<<512, 256, 0, stream>>>(x, xh);

  // node blocks: 27 batched GEMMs [256,512]x[512,512] -> fu (pre-BN)
  k_gemm<<<dim3(16, 4, 27), 256, 0, stream>>>(xh, 512, 0, nbWh, 512, (long)512 * 512,
                                              nullptr, nullptr, fu, 512, (long)256 * 512, 256, 512, 1);
  k_bn_stats<<<27, 256, 0, stream>>>(fu, bnm, bnr);
  k_bn_apply<<<13824, 256, 0, stream>>>(fu, bnm, bnr, nbbnw, nbbnb);
  k_fv_gap<<<216, 256, 0, stream>>>(fu, fv);

  // FAM k/v (v stored token-transposed for WMMA B use)
  k_gemm<<<dim3(16, 2, 1), 256, 0, stream>>>(xh, 512, 0, famkh, 512, 0,
                                             famkb, nullptr, kx, 256, 0, 256, 512, 1);
  k_gemm<<<dim3(16, 4, 1), 256, 0, stream>>>(xh, 512, 0, famvh, 512, 0,
                                             famvb, nullptr, vxT, 512, 0, 256, 512, 2);
  k_fam<<<108, 256, 0, stream>>>(fu, famqh, famqb, kx, vxT, feat);

  // ARM Q/K/V
  k_gemm<<<dim3(432, 2, 1), 256, 0, stream>>>(feat, 512, 0, armqh, 512, 0,
                                              armqb, nullptr, Qb, 256, 0, 6912, 512, 1);
  k_gemm<<<dim3(432, 2, 1), 256, 0, stream>>>(feat, 512, 0, armkh, 512, 0,
                                              armkb, nullptr, Kb, 256, 0, 6912, 512, 1);
  k_gemm<<<dim3(432, 4, 1), 256, 0, stream>>>(feat, 512, 0, armvh, 512, 0,
                                              armvb, nullptr, VTb, 512, 0, 6912, 512, 2);

  // fused ARM attention + edge projection + token-mean + BN partials
  k_arm_edge<<<dim3(729, 4), 256, 0, stream>>>(Qb, Kb, VTb, epwh, epb, tbuf, stat);
  k_gem_bn<<<730, 256, 0, stream>>>(tbuf, stat, gemw, gemb, fe, fe16);

  // 2 GatedGNN layers
  for (int l = 0; l < 2; ++l) {
    long wo = (long)l * 512 * 512;
    k_fvh<<<224, 256, 0, stream>>>(fv, fvh);
    k_gemm<<<dim3(7, 4, 1), 256, 0, stream>>>(fvh, 512, 0, gnnah + wo, 512, 0,
                                              nullptr, vi, nullptr, 512, 0, 108, 512, 0);
    k_gemm<<<dim3(7, 4, 1), 256, 0, stream>>>(fvh, 512, 0, gnnbh + wo, 512, 0,
                                              nullptr, vj, nullptr, 512, 0, 108, 512, 0);
    k_gemm<<<dim3(7, 4, 1), 256, 0, stream>>>(fvh, 512, 0, gnnuh + wo, 512, 0,
                                              nullptr, ui, nullptr, 512, 0, 108, 512, 0);
    k_gemm<<<dim3(7, 4, 1), 256, 0, stream>>>(fvh, 512, 0, gnnvh + wo, 512, 0,
                                              nullptr, uj, nullptr, 512, 0, 108, 512, 0);
    k_gemm<<<dim3(183, 4, 1), 256, 0, stream>>>(fe16, 512, 0, gnneh + wo, 512, 0,
                                                nullptr, el, nullptr, 512, 0, 2916, 512, 0);
    k_edge_upd<<<729, 256, 0, stream>>>(vi, vj, el, gbnew + l * 729, gbneb + l * 729, fe, fe16);
    k_agg<<<108, 256, 0, stream>>>(fe, ui, uj, agg);
    k_node_upd<<<27, 256, 0, stream>>>(agg, gbnvw + l * 27, gbnvb + l * 27, fv);
  }

  k_sc<<<164, 64, 0, stream>>>(fv, mansc, subsc, (float*)d_out);
}
